// DualPathRNN_64553358458914
// MI455X (gfx1250) — matrix operand order repaired and checked by reading, not compiled, not run on hardware
//
#include <hip/hip_runtime.h>
#include <cstddef>
#include <cstdint>

typedef __attribute__((ext_vector_type(16))) _Float16 v16h;
typedef __attribute__((ext_vector_type(8)))  _Float16 v8h_t;
typedef __attribute__((ext_vector_type(8)))  float    v8f;

namespace dpr {
constexpr int B_ = 4, C_ = 64, T_ = 128, Q_ = 128, K_ = 8, H_ = 128;
constexpr int N_   = B_ * T_;        // 512 sequences
constexpr int L_   = Q_ - K_ + 1;    // 121
constexpr int DIN_ = C_ * K_;        // 512
constexpr int M1_  = N_ * L_;        // 61952 (divisible by 128)
constexpr int TH2_ = 136;            // padded time length for conv-transpose unfold
constexpr int NCH2_ = 2 * H_;        // 256
constexpr int KD2_  = NCH2_ * K_;    // 2048
constexpr int M2_   = N_ * Q_;       // 65536

static_assert(M1_ % 128 == 0, "M1 tiling");
static_assert(M2_ % 128 == 0, "M2 tiling");

// ---------------- workspace layout (bytes, 256-aligned) ----------------
constexpr size_t OFF_XLN  = 0;                         // f16 [512][64][128]   = 8,388,608 B
constexpr size_t OFF_HP2  = OFF_XLN  + 8388608;        // f16 [512][256][136]  = 35,651,584 B
constexpr size_t OFF_WTF  = OFF_HP2  + 35651584;       // f16 [512][512]       = 524,288 B
constexpr size_t OFF_WTB  = OFF_WTF  + 524288;         // f16 [512][512]
constexpr size_t OFF_B2   = OFF_WTB  + 524288;         // f16 [64][2048]       = 262,144 B
constexpr size_t OFF_U    = OFF_B2   + 262144;         // f32 [61952][512]     = 126,877,696 B
constexpr size_t OFF_Y    = OFF_U    + 126877696;      // f32 [65536][64]      = 16,777,216 B
} // namespace dpr

using namespace dpr;

// ---------------- LayerNorm over channels -> f16 [n][c][q] ----------------
__global__ __launch_bounds__(256) void dpr_ln_kernel(
    const float* __restrict__ f, const float* __restrict__ gamma,
    const float* __restrict__ beta, _Float16* __restrict__ xln)
{
    int tid = blockIdx.x * 256 + threadIdx.x;          // < N_*Q_ = 65536
    int q = tid & (Q_ - 1);
    int n = tid >> 7;
    int b = n >> 7;
    int t = n & (T_ - 1);
    size_t base = (size_t)b * (C_ * T_ * Q_) + (size_t)t * Q_ + q;
    float s = 0.f, ss = 0.f;
    for (int c = 0; c < C_; ++c) {
        float x = f[base + (size_t)c * (T_ * Q_)];
        s += x; ss += x * x;
    }
    float mu  = s * (1.0f / C_);
    float var = ss * (1.0f / C_) - mu * mu;
    float rs  = rsqrtf(var + 1e-5f);
    size_t ob = (size_t)n * (C_ * Q_) + q;
    for (int c = 0; c < C_; ++c) {
        float x = f[base + (size_t)c * (T_ * Q_)];
        xln[ob + (size_t)c * Q_] = (_Float16)((x - mu) * rs * gamma[c] + beta[c]);
    }
}

// ---------------- zero fill (hp2 padding) ----------------
__global__ __launch_bounds__(256) void dpr_zero_u32(uint32_t* __restrict__ p, int n)
{
    int i = blockIdx.x * 256 + threadIdx.x;
    if (i < n) p[i] = 0u;
}

// ---------------- weight prep: transpose + f16, conv kernel flip ----------------
__global__ __launch_bounds__(256) void dpr_prep_weights(
    const float* __restrict__ wf, const float* __restrict__ wb,
    const float* __restrict__ cw,
    _Float16* __restrict__ wtf, _Float16* __restrict__ wtb, _Float16* __restrict__ b2)
{
    int tid = blockIdx.x * 256 + threadIdx.x;
    if (tid < 262144) {                                  // wtf[col][d] = wf[d][col]
        int col = tid >> 9, d = tid & 511;
        wtf[tid] = (_Float16)wf[(size_t)d * 512 + col];
    } else if (tid < 524288) {
        int i = tid - 262144;
        int col = i >> 9, d = i & 511;
        wtb[i] = (_Float16)wb[(size_t)d * 512 + col];
    } else if (tid < 655360) {                           // b2[c][j*8+k'] = ct_w[j][c][7-k']
        int i = tid - 524288;
        int c = i >> 11, d = i & 2047;
        int j = d >> 3, kk = d & 7;
        b2[i] = (_Float16)cw[((size_t)j * 64 + c) * 8 + (7 - kk)];
    }
}

// ---------------- WMMA fragment loads ----------------
// A 16x32 f16 layout: lane = (m & 15) + 16*khalf ; elements 0..7 -> K = khalf*8+j,
// elements 8..15 -> K = 16 + khalf*8 + (j-8).  With d = ch*8 + k and A[(n,l)][d] =
// src[n][ch][l+k], each 8-run is 8 contiguous time samples of one channel.
__device__ __forceinline__ v16h dpr_load_a(
    const _Float16* __restrict__ src, int n, int l, int kc, int khalf, int NCH, int Tsrc)
{
    const int ch = kc * 4 + khalf;
    const _Float16* p0 = src + ((size_t)n * NCH + ch) * Tsrc + l;
    const _Float16* p1 = p0 + 2 * (size_t)Tsrc;
    v16h a;
#pragma unroll
    for (int j = 0; j < 8; ++j) { a[j] = p0[j]; a[j + 8] = p1[j]; }
    return a;
}

// B 32x16 f16 layout: lane = n + 16*khalf ; elements 0..7 -> K = khalf*8 + j, elements 8..15 -> K = 16 + khalf*8 + (j-8).
// B K-chunk staged in LDS as [64 local cols][32 k] -> per lane two runs of 8 contiguous f16, 16 apart.
__device__ __forceinline__ v16h dpr_load_b_lds(
    const _Float16* bsh, int lcol, int khalf)
{
    const v8h_t* p = (const v8h_t*)(bsh + lcol * 32 + khalf * 8);
    v8h_t lo = p[0];
    v8h_t hi = p[2];
    return __builtin_shufflevector(lo, hi, 0,1,2,3,4,5,6,7,8,9,10,11,12,13,14,15);
}

// ---------------- sliding-window WMMA GEMM ----------------
// out[M][Nout] f32 = unfold(src)[M][Kdim] @ Bm^T ;  row r -> (n = r/Lout, l = r%Lout)
// Block: 8 waves (4 row x 2 col) -> 128x64 tile; wave: 2x2 of 16x16 WMMA tiles.
// B K-chunks are staged cooperatively into LDS (one 4 KB chunk per step).
__global__ __launch_bounds__(256) void dpr_gemm_unfold_wmma(
    const _Float16* __restrict__ src, int NCH, int Tsrc, int Lout,
    const _Float16* __restrict__ Bm, int Kdim, int Nout,
    float* __restrict__ out)
{
    __shared__ _Float16 bsh[64 * 32];                  // one K-chunk of the B tile

    const int tid   = threadIdx.x;
    const int lane  = tid & 31;
    const int wave  = tid >> 5;
    const int rw    = wave & 3;
    const int cw    = wave >> 2;
    const int khalf = lane >> 4;
    const int lm    = lane & 15;

    const int rowt0 = blockIdx.x * 128 + rw * 32;
    const int colb  = blockIdx.y * 64;                 // block column base (global)
    const int lcol0 = cw * 32 + lm;                    // local column of subtile 0
    const int lcol1 = lcol0 + 16;

    const int row0 = rowt0 + lm;
    const int row1 = row0 + 16;

    const int n0 = row0 / Lout, l0 = row0 - n0 * Lout;
    const int n1 = row1 / Lout, l1 = row1 - n1 * Lout;

    // cooperative B staging: 256 threads x 8 halfs = 64x32 chunk
    const int scol = tid >> 2;                         // 0..63 local col
    const int sseg = (tid & 3) * 8;                    // 0,8,16,24 within chunk
    const _Float16* bsrc = Bm + (size_t)(colb + scol) * Kdim + sseg;
    _Float16* bdst = &bsh[scol * 32 + sseg];

    v8f c00 = {}, c01 = {}, c10 = {}, c11 = {};
    const int nk = Kdim >> 5;
    for (int kc = 0; kc < nk; ++kc) {
        __syncthreads();
        *(v8h_t*)bdst = *(const v8h_t*)(bsrc + (size_t)kc * 32);
        __syncthreads();

        if (kc + 1 < nk) {                             // prefetch next A chunk
            const int ch = (kc + 1) * 4 + khalf;
            __builtin_prefetch(src + ((size_t)n0 * NCH + ch) * Tsrc + l0, 0, 1);
            __builtin_prefetch(src + ((size_t)n1 * NCH + ch) * Tsrc + l1, 0, 1);
        }

        v16h a0 = dpr_load_a(src, n0, l0, kc, khalf, NCH, Tsrc);
        v16h a1 = dpr_load_a(src, n1, l1, kc, khalf, NCH, Tsrc);
        v16h b0 = dpr_load_b_lds(bsh, lcol0, khalf);
        v16h b1 = dpr_load_b_lds(bsh, lcol1, khalf);
        c00 = __builtin_amdgcn_wmma_f32_16x16x32_f16(false, a0, false, b0, (short)0, c00, false, false);
        c01 = __builtin_amdgcn_wmma_f32_16x16x32_f16(false, a0, false, b1, (short)0, c01, false, false);
        c10 = __builtin_amdgcn_wmma_f32_16x16x32_f16(false, a1, false, b0, (short)0, c10, false, false);
        c11 = __builtin_amdgcn_wmma_f32_16x16x32_f16(false, a1, false, b1, (short)0, c11, false, false);
    }
    // C/D 16x16 f32: VGPR v, lanes 0-15 -> M=v ; lanes 16-31 -> M=v+8 ; N = lane&15
    const int rb0 = rowt0 + khalf * 8;
    const int rb1 = rb0 + 16;
    const int gcol0 = colb + lcol0;
    const int gcol1 = colb + lcol1;
#pragma unroll
    for (int v = 0; v < 8; ++v) {
        out[(size_t)(rb0 + v) * Nout + gcol0] = c00[v];
        out[(size_t)(rb0 + v) * Nout + gcol1] = c01[v];
        out[(size_t)(rb1 + v) * Nout + gcol0] = c10[v];
        out[(size_t)(rb1 + v) * Nout + gcol1] = c11[v];
    }
}

// ---------------- SRU recurrence + gates -> hp2 (padded, channel-major) ----------------
__global__ __launch_bounds__(256) void dpr_sru_scan(
    const float* __restrict__ U, const float* __restrict__ bias,
    _Float16* __restrict__ hp2, int reverse)
{
    int tid = blockIdx.x * 256 + threadIdx.x;          // < N_*H_ = 65536
    int h = tid & (H_ - 1);
    int n = tid >> 7;
    float bf = bias[h], br = bias[H_ + h];
    float c = 0.f;
    _Float16* dst = hp2 + ((size_t)n * NCH2_ + (reverse ? H_ + h : h)) * TH2_;
    if (!reverse) {
        for (int l = 0; l < L_; ++l) {
            const float* row = U + (size_t)(n * L_ + l) * (4 * H_);
            float xt = row[h], fp = row[H_ + h], rp = row[2 * H_ + h], xp = row[3 * H_ + h];
            float fg = 1.f / (1.f + expf(-(fp + bf)));
            float rg = 1.f / (1.f + expf(-(rp + br)));
            c = fg * c + (1.f - fg) * xt;
            dst[l + (K_ - 1)] = (_Float16)(rg * tanhf(c) + (1.f - rg) * xp);
        }
    } else {
        for (int l = L_ - 1; l >= 0; --l) {
            const float* row = U + (size_t)(n * L_ + l) * (4 * H_);
            float xt = row[h], fp = row[H_ + h], rp = row[2 * H_ + h], xp = row[3 * H_ + h];
            float fg = 1.f / (1.f + expf(-(fp + bf)));
            float rg = 1.f / (1.f + expf(-(rp + br)));
            c = fg * c + (1.f - fg) * xt;
            dst[l + (K_ - 1)] = (_Float16)(rg * tanhf(c) + (1.f - rg) * xp);
        }
    }
}

// ---------------- bias + residual into [B,C,T,Q] ----------------
__global__ __launch_bounds__(256) void dpr_final_add(
    const float* __restrict__ feat, const float* __restrict__ ctb,
    const float* __restrict__ ytmp, float* __restrict__ out)
{
    int idx = blockIdx.x * 256 + threadIdx.x;          // < 4,194,304
    int q = idx & 127;
    int t = (idx >> 7) & 127;
    int c = (idx >> 14) & 63;
    int b = idx >> 20;
    int r = (b * T_ + t) * Q_ + q;
    out[idx] = feat[idx] + ctb[c] + ytmp[(size_t)r * C_ + c];
}

extern "C" void kernel_launch(void* const* d_in, const int* in_sizes, int n_in,
                              void* d_out, int out_size, void* d_ws, size_t ws_size,
                              hipStream_t stream)
{
    (void)in_sizes; (void)n_in; (void)out_size; (void)ws_size;
    const float* features = (const float*)d_in[0];
    const float* ln_gamma = (const float*)d_in[1];
    const float* ln_beta  = (const float*)d_in[2];
    const float* w_fwd    = (const float*)d_in[3];
    const float* b_fwd    = (const float*)d_in[4];
    const float* w_bwd    = (const float*)d_in[5];
    const float* b_bwd    = (const float*)d_in[6];
    const float* ct_w     = (const float*)d_in[7];
    const float* ct_b     = (const float*)d_in[8];
    float* out = (float*)d_out;

    char* ws = (char*)d_ws;
    _Float16* xln = (_Float16*)(ws + OFF_XLN);
    _Float16* hp2 = (_Float16*)(ws + OFF_HP2);
    _Float16* wtf = (_Float16*)(ws + OFF_WTF);
    _Float16* wtb = (_Float16*)(ws + OFF_WTB);
    _Float16* b2  = (_Float16*)(ws + OFF_B2);
    float*    U   = (float*)(ws + OFF_U);
    float*    ytmp= (float*)(ws + OFF_Y);

    // 1) LayerNorm -> xln
    dpr_ln_kernel<<<(N_ * Q_) / 256, 256, 0, stream>>>(features, ln_gamma, ln_beta, xln);

    // 2) zero hp2 padding (f16 count / 2 = u32 count)
    {
        int nu32 = (N_ * NCH2_ * TH2_) / 2;
        dpr_zero_u32<<<(nu32 + 255) / 256, 256, 0, stream>>>((uint32_t*)hp2, nu32);
    }

    // 3) weight prep
    dpr_prep_weights<<<(655360) / 256, 256, 0, stream>>>(w_fwd, w_bwd, ct_w, wtf, wtb, b2);

    // 4) forward direction: GEMM1 -> U, then scan -> hp2[:, 0:128, :]
    dpr_gemm_unfold_wmma<<<dim3(M1_ / 128, 512 / 64), 256, 0, stream>>>(
        xln, C_, Q_, L_, wtf, DIN_, 4 * H_, U);
    dpr_sru_scan<<<(N_ * H_) / 256, 256, 0, stream>>>(U, b_fwd, hp2, 0);

    // 5) backward direction (reuses U): GEMM1 -> U, scan -> hp2[:, 128:256, :]
    dpr_gemm_unfold_wmma<<<dim3(M1_ / 128, 512 / 64), 256, 0, stream>>>(
        xln, C_, Q_, L_, wtb, DIN_, 4 * H_, U);
    dpr_sru_scan<<<(N_ * H_) / 256, 256, 0, stream>>>(U, b_bwd, hp2, 1);

    // 6) ConvTranspose as sliding-window GEMM over hp2 -> ytmp [65536][64]
    dpr_gemm_unfold_wmma<<<dim3(M2_ / 128, 64 / 64), 256, 0, stream>>>(
        hp2, NCH2_, TH2_, Q_, b2, KD2_, C_, ytmp);

    // 7) bias + residual -> d_out
    dpr_final_add<<<(B_ * C_ * T_ * Q_) / 256, 256, 0, stream>>>(features, ct_b, ytmp, out);
}
